// PercieverAttention_58609123721905
// MI455X (gfx1250) — matrix operand order repaired and checked by reading, not compiled, not run on hardware
//
#include <hip/hip_runtime.h>
#include <hip/hip_bf16.h>

typedef __attribute__((ext_vector_type(16))) __bf16 bf16x16;
typedef __attribute__((ext_vector_type(8)))  __bf16 bf16x8;
typedef __attribute__((ext_vector_type(8)))  float  f32x8;

#define NL    64      // latent rows per (b,t)
#define DIMC  128     // model dim (K of projections)
#define DH    256     // per-head dim (2048/8)
#define NHEAD 8
#define NKV   1088    // 1024 tensor rows + 64 latent rows
#define JT    64      // kv tile rows
#define NTILE 17      // NKV / JT
#define QSCALE 0.08838834764831845f  // 128^-0.5

// ---- LDS layout (bytes), total 287488 < 320KB/WGP ----
#define OFF_Q   0              // bf16 [64][256]   32768
#define OFF_WA  32768          // bf16 [256][128]T 65536  (WqT / WkT / WoT)
#define OFF_WB  98304          // bf16 [256][128]T 65536  (WvT)
#define OFF_X0  163840         // bf16 [64][128]   16384  (kv tile buf 0)
#define OFF_X1  180224         // bf16 [64][128]   16384  (kv tile buf 1)
#define OFF_K   196608         // bf16 [64][256]   32768  (k tile; reused as O)
#define OFF_VT  229376         // bf16 [256][64]   32768  (v tile transposed)
#define OFF_S   262144         // f32  [64][64]    16384  (sim tile)
#define OFF_P   278528         // bf16 [64][64]     8192  (probs tile)
#define OFF_ST  286720         // f32  sM[64], sL[64], sC[64]
#define SMEM_BYTES 287488

// ws layout: [0, 4MB) fp32 head contributions; [4MB, ...) bf16 packed kv input
#define WS_KV_OFF (16 * NHEAD * NL * DIMC * 4)   // 4194304 bytes

__device__ inline f32x8 zero8() {
  f32x8 z;
#pragma unroll
  for (int i = 0; i < 8; ++i) z[i] = 0.0f;
  return z;
}

// A fragment: row-major [M][K] source, 16x32 bf16 tile.
// lanes 0-15 hold K {0..7,16..23}, lanes 16-31 hold K {8..15,24..31} (ISA 7.12.2).
__device__ inline bf16x16 load_a(const __bf16* base, int ld, int row, int k0, int half) {
  const __bf16* p = base + row * ld + k0 + 8 * half;
  bf16x8 lo = *(const bf16x8*)p;
  bf16x8 hi = *(const bf16x8*)(p + 16);
  bf16x16 r;
#pragma unroll
  for (int i = 0; i < 8; ++i) { r[i] = lo[i]; r[i + 8] = hi[i]; }
  return r;
}

// B fragment: source stored as [N][K] (transposed), so lane (= column n) reads
// 2x16 bytes. lanes 0-15: K {0..7,16..23}, lanes 16-31: K {8..15,24..31} of the tile.
__device__ inline bf16x16 load_bT(const __bf16* baseT, int ldk, int col, int k0, int half) {
  const __bf16* p = baseT + col * ldk + k0 + 8 * half;
  bf16x8 lo = *(const bf16x8*)p;
  bf16x8 hi = *(const bf16x8*)(p + 16);
  bf16x16 r;
#pragma unroll
  for (int i = 0; i < 8; ++i) { r[i] = lo[i]; r[i + 8] = hi[i]; }
  return r;
}

__device__ inline f32x8 wmma_bf16(bf16x16 a, bf16x16 b, f32x8 c) {
  return __builtin_amdgcn_wmma_f32_16x16x32_bf16(false, a, false, b, (short)0, c, false, false);
}

// Async copy of one 16KB tile (64x128 bf16) global->LDS, spread over 8 waves.
// Each wave issues 4 x global_load_async_to_lds_b128 (32 lanes x 16B = 512B each).
__device__ inline void async_tile_16k(uint32_t lds_off, const __bf16* gsrc,
                                      int wave, int lane) {
  uint32_t lo = lds_off + (uint32_t)(wave * 2048 + lane * 16);
  const char* g = (const char*)gsrc + wave * 2048 + lane * 16;
#pragma unroll
  for (int i = 0; i < 4; ++i) {
    asm volatile("global_load_async_to_lds_b128 %0, %1, off"
                 :: "v"(lo + (uint32_t)(i * 512)), "v"(g + i * 512)
                 : "memory");
  }
}

__device__ inline uint32_t lds_offset_of(const void* p) {
  return (uint32_t)(uintptr_t)p;   // low 32 bits of flat LDS pointer = LDS byte offset
}

// ---- prep: pack concat(tensor, latents) to bf16 once per (b,t) ----
__global__ __launch_bounds__(256) void pack_kv_bf16(const float* __restrict__ tensor,
                                                    const float* __restrict__ latents,
                                                    __bf16* __restrict__ kvin) {
  int idx = blockIdx.x * 256 + threadIdx.x;     // 16*1088*128 = 2228224 total
  int bt = idx / (NKV * DIMC);
  int r  = idx - bt * (NKV * DIMC);
  int j = r >> 7, c = r & 127;
  float v = (j < 1024) ? tensor[(size_t)bt * 1024 * DIMC + (size_t)j * DIMC + c]
                       : latents[(size_t)bt * NL * DIMC + (size_t)(j - 1024) * DIMC + c];
  kvin[idx] = (__bf16)v;
}

__global__ __launch_bounds__(256) void perceiver_attn_fused(
    const __bf16* __restrict__ kvin,
    const float* __restrict__ Wq, const float* __restrict__ Wkv,
    const float* __restrict__ Wo, float* __restrict__ head_out) {
  extern __shared__ char smem[];
  __bf16* sQ  = (__bf16*)(smem + OFF_Q);
  __bf16* sWa = (__bf16*)(smem + OFF_WA);
  __bf16* sWb = (__bf16*)(smem + OFF_WB);
  __bf16* sX0 = (__bf16*)(smem + OFF_X0);
  __bf16* sX1 = (__bf16*)(smem + OFF_X1);
  __bf16* sK  = (__bf16*)(smem + OFF_K);   // later reused as normalized O
  __bf16* sVt = (__bf16*)(smem + OFF_VT);
  float*  sS  = (float*)(smem + OFF_S);
  __bf16* sP  = (__bf16*)(smem + OFF_P);
  float*  sM  = (float*)(smem + OFF_ST);
  float*  sL  = sM + 64;
  float*  sC  = sL + 64;

  const int tid  = threadIdx.x;
  const int wave = tid >> 5;
  const int lane = tid & 31;
  const int l    = lane & 15;
  const int half = lane >> 4;
  const int bt   = blockIdx.x >> 3;   // 0..15
  const int h    = blockIdx.x & 7;    // head

  const __bf16* kvbt = kvin + (size_t)bt * NKV * DIMC;
  const uint32_t offX0 = lds_offset_of(sX0);
  const uint32_t offX1 = lds_offset_of(sX1);

  // ---- prologue: async-stage latents rows (for q-proj) into X0, kv tile 0 into X1 ----
  async_tile_16k(offX0, kvbt + 1024 * DIMC, wave, lane);
  async_tile_16k(offX1, kvbt, wave, lane);

  // stage WqT slice (fp32 -> bf16, transposed [e][c]); init softmax stats
  for (int idx = tid; idx < DIMC * DH; idx += 256) {
    int c = idx >> 8, e = idx & 255;                   // contiguous global reads over e
    sWa[e * DIMC + c] = (__bf16)Wq[c * 2048 + h * DH + e];
  }
  if (tid < 64) { sM[tid] = -1e30f; sL[tid] = 0.0f; }
  asm volatile("s_wait_asynccnt 0x4" ::: "memory");    // X0 (latents) landed; tile0 in flight
  __syncthreads();

  const int mt = wave >> 1;        // row tile 0..3 owned by this wave
  const int m0 = mt * 16;

  // ---- Phase 1: q = latents @ Wq_h, scaled, -> sQ (bf16) ----
  for (int t = 0; t < 8; ++t) {
    int n0 = ((wave & 1) * 8 + t) * 16;
    f32x8 acc = zero8();
#pragma unroll
    for (int kk = 0; kk < 4; ++kk) {
      bf16x16 a = load_a(sX0, DIMC, m0 + l, kk * 32, half);
      bf16x16 b = load_bT(sWa, DIMC, n0 + l, kk * 32, half);
      acc = wmma_bf16(a, b, acc);
    }
#pragma unroll
    for (int r = 0; r < 8; ++r)
      sQ[(m0 + r + 8 * half) * DH + n0 + l] = (__bf16)(acc[r] * QSCALE);
  }
  __syncthreads();

  // ---- stage WkT / WvT slices ----
  for (int idx = tid; idx < DIMC * DH; idx += 256) {
    int c = idx >> 8, e = idx & 255;
    sWa[e * DIMC + c] = (__bf16)Wkv[c * 4096 + h * DH + e];
    sWb[e * DIMC + c] = (__bf16)Wkv[c * 4096 + 2048 + h * DH + e];
  }
  __syncthreads();

  f32x8 o_acc[8];
#pragma unroll
  for (int t = 0; t < 8; ++t) o_acc[t] = zero8();

  // ---- stream kv rows in tiles of 64, double-buffered async copies ----
  for (int jt = 0; jt < NTILE; ++jt) {
    const __bf16* cur = (jt & 1) ? sX0 : sX1;          // tile0 -> X1, tile1 -> X0, ...
    if (jt + 1 < NTILE) {
      uint32_t dst = (jt & 1) ? offX1 : offX0;         // next tile -> other buffer
      async_tile_16k(dst, kvbt + (size_t)(jt + 1) * JT * DIMC, wave, lane);
      asm volatile("s_wait_asynccnt 0x4" ::: "memory"); // current tile done, next in flight
    } else {
      asm volatile("s_wait_asynccnt 0x0" ::: "memory");
    }
    __syncthreads();

    // (b) k_tile = X @ Wk_h -> sK [j][d];  v_tile = X @ Wv_h -> sVt [d][j]
    for (int t = 0; t < 8; ++t) {
      int n0 = ((wave & 1) * 8 + t) * 16;
      f32x8 ak = zero8(), av = zero8();
#pragma unroll
      for (int kk = 0; kk < 4; ++kk) {
        bf16x16 a  = load_a(cur, DIMC, m0 + l, kk * 32, half);
        bf16x16 bk = load_bT(sWa, DIMC, n0 + l, kk * 32, half);
        bf16x16 bv = load_bT(sWb, DIMC, n0 + l, kk * 32, half);
        ak = wmma_bf16(a, bk, ak);
        av = wmma_bf16(a, bv, av);
      }
#pragma unroll
      for (int r = 0; r < 8; ++r) {
        sK[(m0 + r + 8 * half) * DH + n0 + l]   = (__bf16)ak[r];
        sVt[(n0 + l) * JT + m0 + 8 * half + r]  = (__bf16)av[r];
      }
    }
    __syncthreads();

    // (c) sim tile [64 x 64] = q @ k^T (contraction over d=256)
    for (int t = 0; t < 2; ++t) {
      int n0 = ((wave & 1) * 2 + t) * 16;
      f32x8 acc = zero8();
#pragma unroll
      for (int kk = 0; kk < 8; ++kk) {
        bf16x16 a = load_a(sQ, DH, m0 + l, kk * 32, half);
        bf16x16 b = load_bT(sK, DH, n0 + l, kk * 32, half);   // sK is [j][d] == B^T layout
        acc = wmma_bf16(a, b, acc);
      }
#pragma unroll
      for (int r = 0; r < 8; ++r)
        sS[(m0 + r + 8 * half) * JT + n0 + l] = acc[r];
    }
    __syncthreads();

    // (d) online softmax row update (one thread per q row)
    if (tid < 64) {
      int i = tid;
      float mold = sM[i], mx = mold;
      for (int j = 0; j < JT; ++j) mx = fmaxf(mx, sS[i * JT + j]);
      float corr = __expf(mold - mx);
      float sum = 0.0f;
      for (int j = 0; j < JT; ++j) {
        float p = __expf(sS[i * JT + j] - mx);
        sP[i * JT + j] = (__bf16)p;
        sum += p;
      }
      sL[i] = sL[i] * corr + sum;
      sM[i] = mx;
      sC[i] = corr;
    }
    __syncthreads();

    // (e) rescale accumulators, O += P @ V
    {
      float cr[8];
#pragma unroll
      for (int r = 0; r < 8; ++r) cr[r] = sC[m0 + 8 * half + r];
      for (int t = 0; t < 8; ++t) {
        int n0 = ((wave & 1) * 8 + t) * 16;
#pragma unroll
        for (int r = 0; r < 8; ++r) o_acc[t][r] *= cr[r];
#pragma unroll
        for (int kk = 0; kk < 2; ++kk) {
          bf16x16 a = load_a(sP, JT, m0 + l, kk * 32, half);
          bf16x16 b = load_bT(sVt, JT, n0 + l, kk * 32, half); // sVt is [d][j] == B^T layout
          o_acc[t] = wmma_bf16(a, b, o_acc[t]);
        }
      }
    }
    __syncthreads();
  }

  // ---- epilogue: normalize O -> bf16 (reuse sK), stage WoT ----
  {
    float inv[8];
#pragma unroll
    for (int r = 0; r < 8; ++r) inv[r] = 1.0f / sL[m0 + 8 * half + r];
    for (int t = 0; t < 8; ++t) {
      int n0 = ((wave & 1) * 8 + t) * 16;
#pragma unroll
      for (int r = 0; r < 8; ++r)
        sK[(m0 + r + 8 * half) * DH + n0 + l] = (__bf16)(o_acc[t][r] * inv[r]);
    }
  }
  for (int idx = tid; idx < DH * DIMC; idx += 256) {
    int e = idx >> 7, c = idx & 127;                  // contiguous global reads over c
    sWa[c * DH + e] = (__bf16)Wo[(size_t)(h * DH + e) * DIMC + c];
  }
  __syncthreads();

  // ---- contribution = O_h @ Wo_h -> head_out[bt][h][64][128] ----
  {
    float* outp = head_out + (size_t)(bt * NHEAD + h) * NL * DIMC;
    for (int t = 0; t < 4; ++t) {
      int n0 = ((wave & 1) * 4 + t) * 16;
      f32x8 acc = zero8();
#pragma unroll
      for (int kk = 0; kk < 8; ++kk) {
        bf16x16 a = load_a(sK, DH, m0 + l, kk * 32, half);
        bf16x16 b = load_bT(sWa, DH, n0 + l, kk * 32, half);
        acc = wmma_bf16(a, b, acc);
      }
#pragma unroll
      for (int r = 0; r < 8; ++r)
        outp[(m0 + r + 8 * half) * DIMC + n0 + l] = acc[r];
    }
  }
}

// Deterministic sum over the 8 head contributions.
__global__ __launch_bounds__(256) void reduce_heads(const float* __restrict__ hws,
                                                    float* __restrict__ out) {
  int idx = blockIdx.x * 256 + threadIdx.x;   // 0..131071
  int bt = idx >> 13;                         // / (64*128)
  int rc = idx & 8191;
  float s = 0.0f;
#pragma unroll
  for (int h = 0; h < NHEAD; ++h)
    s += hws[(size_t)(bt * NHEAD + h) * 8192 + rc];
  out[idx] = s;
}

extern "C" void kernel_launch(void* const* d_in, const int* in_sizes, int n_in,
                              void* d_out, int out_size, void* d_ws, size_t ws_size,
                              hipStream_t stream) {
  (void)in_sizes; (void)n_in; (void)out_size; (void)ws_size;
  const float* tensor  = (const float*)d_in[0];
  const float* latents = (const float*)d_in[1];
  const float* Wq      = (const float*)d_in[2];
  const float* Wkv     = (const float*)d_in[3];
  const float* Wo      = (const float*)d_in[4];
  float* out = (float*)d_out;
  float*  hws  = (float*)d_ws;                          // 4 MB head contributions
  __bf16* kvin = (__bf16*)((char*)d_ws + WS_KV_OFF);    // 4.45 MB packed bf16 kv input

  hipFuncSetAttribute((const void*)perceiver_attn_fused,
                      hipFuncAttributeMaxDynamicSharedMemorySize, SMEM_BYTES);

  pack_kv_bf16<<<(16 * NKV * DIMC) / 256, 256, 0, stream>>>(tensor, latents, kvin);
  perceiver_attn_fused<<<16 * NHEAD, 256, SMEM_BYTES, stream>>>(
      kvin, Wq, Wkv, Wo, hws);
  reduce_heads<<<(16 * 64 * 128) / 256, 256, 0, stream>>>(hws, out);
}
